// cw_vit_13503377178902
// MI455X (gfx1250) — compile-verified, matrix operand order checked by reading, not run on hardware
//
#include <hip/hip_runtime.h>
#include <math.h>
#include <stdint.h>

// ---------------- types ----------------
typedef __attribute__((ext_vector_type(16))) __bf16 v16bf;
typedef __attribute__((ext_vector_type(8)))  __bf16 v8bf;
typedef __attribute__((ext_vector_type(8)))  float  v8f;

#define DEVINL static __device__ __forceinline__

// ---------------- problem dims ----------------
#define IMGE 15
#define MIDP 7
#define NRINGS 7
#define NPATCH 56
#define BATCH 512
#define CHN 256
#define DIM 512
#define HEADS 8
#define DH 64
#define DEPTH 6
#define MLPD 2048
#define NCLS 16
#define TOKN 57
#define ROWS (BATCH * TOKN)    /* 29184 */
#define EROWS (BATCH * NPATCH) /* 28672 */

// ---------------- constexpr spiral ring tables (replica of _ring) ----------------
struct RingTables {
  int tgt[NRINGS][NPATCH];
  int s1[NRINGS][NPATCH];
  int s2[NRINGS][NPATCH];
  int cnt[NRINGS];
};
constexpr int csign(int v) { return v > 0 ? 1 : (v < 0 ? -1 : 0); }
constexpr RingTables gen_rings() {
  RingTables t{};
  for (int i = 1; i <= NRINGS; ++i) {
    const int a1 = MIDP - i, a2 = MIDP + i, b1 = MIDP - i, b2 = MIDP + i;
    const int E = IMGE;
    int d0 = 0, d1 = 1, l1 = a1, l2 = b1 - 1, start = 0, n = 0;
    while (true) {
      l1 += d0; l2 += d1;
      if (l1 == a1 && l2 == b1) { if (start == 1) break; start = 1; }
      if (l1 > a2)      { d0 = 0;  d1 = -1; l1 -= 1; continue; }
      else if (l2 < b1) { d0 = -1; d1 = 0;  l2 += 1; continue; }
      else if (l2 > b2) { d0 = 1;  d1 = 0;  l2 -= 1; continue; }
      if (l1 < 0 || l1 > E || l2 < 0 || l2 > E) continue;
      const int m1 = (l1 == MIDP) ? 0 : -csign(l1 - MIDP);
      const int m2 = (l2 == MIDP) ? 0 : -csign(l2 - MIDP);
      int A = l1 - MIDP; if (A < 0) A = -A;
      int B = l2 - MIDP; if (B < 0) B = -B;
      int v1 = 0, v2 = 0;
      if (A > B)      { v1 = (l1 + m1) * E + l2;      v2 = (l1 + m1) * E + l2 + m2; }
      else if (A < B) { v1 = l1 * E + l2 + m2;        v2 = (l1 + m1) * E + l2 + m2; }
      else            { v1 = (l1 + m1) * E + l2 + m2; v2 = v1; }
      t.tgt[i - 1][n] = l1 * E + l2; t.s1[i - 1][n] = v1; t.s2[i - 1][n] = v2; ++n;
    }
    t.cnt[i - 1] = n;
  }
  return t;
}
__constant__ RingTables g_rings = gen_rings();

// ---------------- WMMA helpers ----------------
// 16-bit A/B fragment: per lane two contiguous 8-halfword chunks at row + hs and row + hs + 16,
// hs = 0 for lanes 0-15, 8 for lanes 16-31 (CDNA5 16-bit A-matrix 16x32 layout).
DEVINL v16bf frag_ld(const __bf16* rowbase, int hs) {
  v8bf a = *(const v8bf*)(rowbase + hs);
  v8bf b = *(const v8bf*)(rowbase + hs + 16);
  v16bf r;
#pragma unroll
  for (int i = 0; i < 8; ++i) { r[i] = a[i]; r[i + 8] = b[i]; }
  return r;
}
DEVINL v8f wmma_bf16(v16bf a, v16bf b, v8f c) {
  return __builtin_amdgcn_wmma_f32_16x16x32_bf16(false, a, false, b, (short)0, c, false, false);
}
DEVINL float gelu_exact(float x) {
  return 0.5f * x * (1.0f + erff(x * 0.7071067811865476f));
}

// ---------------- small utility kernels ----------------
__global__ void k_zero64(uint64_t* p, size_t n) {
  size_t i = (size_t)blockIdx.x * blockDim.x + threadIdx.x;
  if (i < n) p[i] = 0ull;
}

__global__ void k_transpose_bf16(const float* in, __bf16* out, int R, int C) {
  int i = blockIdx.x * 256 + threadIdx.x;
  if (i >= R * C) return;
  int r = i / C, c = i % C;
  out[(size_t)c * R + r] = (__bf16)in[i];
}

__global__ void k_pos(float* POS) {
  int i = blockIdx.x * 256 + threadIdx.x;
  if (i >= TOKN * DIM) return;
  int pp = i >> 9, j = i & 511;
  float ang = (float)pp / powf(10000.0f, (float)(2 * (j / 2)) / 512.0f);
  POS[i] = (j & 1) ? cosf(ang) : sinf(ang);
}

__global__ void k_cls(const float* cls, const float* POS, float* h) {
  int i = blockIdx.x * 256 + threadIdx.x;
  if (i >= BATCH * DIM) return;
  int b = i >> 9, j = i & 511;
  h[(size_t)(b * TOKN) * DIM + j] = cls[j] + POS[j];
}

// ---------------- spiral preprocessing -> patch tokens (bf16) ----------------
__global__ __launch_bounds__(64) void k_spiral(const float* __restrict__ x, __bf16* __restrict__ tok) {
  __shared__ float sb[64 * 225];
  const int pair = blockIdx.x * 64 + threadIdx.x;  // b*256 + c
  const int b = pair >> 8, c = pair & 255;
  float* buf = &sb[threadIdx.x * 225];
  const float* src = x + (size_t)pair * 225;
  for (int i = 0; i < 225; ++i) buf[i] = src[i];
  for (int ring = 0; ring < NRINGS; ++ring) {
    const int n = g_rings.cnt[ring];
    for (int j = 0; j < n; ++j)
      buf[g_rings.tgt[ring][j]] += (buf[g_rings.s1[ring][j]] + buf[g_rings.s2[ring][j]]) * 0.25f;
  }
  for (int pp = 0; pp < NPATCH; ++pp)
    tok[((size_t)(b * NPATCH + pp)) * CHN + c] = (__bf16)buf[g_rings.tgt[NRINGS - 1][pp]];
}

// ---------------- LayerNorm (f32 in -> bf16 out), one wave per row of 512 ----------------
__global__ __launch_bounds__(256) void k_ln(const float* __restrict__ X, const float* __restrict__ G,
                                            const float* __restrict__ Bt, __bf16* __restrict__ Y) {
  const int w = threadIdx.x >> 5, lane = threadIdx.x & 31;
  const int row = blockIdx.x * 8 + w;
  const float4* x4 = (const float4*)(X + (size_t)row * DIM + lane * 16);
  float a[16];
#pragma unroll
  for (int i = 0; i < 4; ++i) {
    float4 t = x4[i];
    a[4 * i] = t.x; a[4 * i + 1] = t.y; a[4 * i + 2] = t.z; a[4 * i + 3] = t.w;
  }
  float s = 0.f, s2 = 0.f;
#pragma unroll
  for (int i = 0; i < 16; ++i) { s += a[i]; s2 += a[i] * a[i]; }
#pragma unroll
  for (int m = 16; m >= 1; m >>= 1) { s += __shfl_xor(s, m, 32); s2 += __shfl_xor(s2, m, 32); }
  const float mu = s * (1.0f / 512.0f);
  const float var = s2 * (1.0f / 512.0f) - mu * mu;
  const float rstd = rsqrtf(var + 1e-5f);
  const float4* g4 = (const float4*)(G + lane * 16);
  const float4* b4 = (const float4*)(Bt + lane * 16);
  float gg[16], bb[16];
#pragma unroll
  for (int i = 0; i < 4; ++i) {
    float4 t = g4[i]; gg[4 * i] = t.x; gg[4 * i + 1] = t.y; gg[4 * i + 2] = t.z; gg[4 * i + 3] = t.w;
    float4 u = b4[i]; bb[4 * i] = u.x; bb[4 * i + 1] = u.y; bb[4 * i + 2] = u.z; bb[4 * i + 3] = u.w;
  }
  v8bf o0, o1;
#pragma unroll
  for (int i = 0; i < 16; ++i) {
    float yv = (a[i] - mu) * rstd * gg[i] + bb[i];
    if (i < 8) o0[i] = (__bf16)yv; else o1[i - 8] = (__bf16)yv;
  }
  __bf16* yr = Y + (size_t)row * DIM + lane * 16;
  *(v8bf*)yr = o0;
  *(v8bf*)(yr + 8) = o1;
}

// ---------------- generic bf16 WMMA GEMM, double-buffered LDS, fused epilogues ----------------
enum { EP_EMBED = 0, EP_QKV = 1, EP_BIAS_RES = 2, EP_GELU = 3 };

struct GemmArgs {
  const __bf16* A;    // [M,K] row-major bf16
  const __bf16* Bt;   // [N,K] row-major bf16 (pre-transposed weights)
  const float* bias;
  const float* res;
  const float* pos;
  float* outF;
  __bf16* outB;
  __bf16* q; __bf16* k; __bf16* vT;
  int N; int K; int mode;
};

__global__ __launch_bounds__(256) void k_gemm(GemmArgs p) {
  __shared__ __bf16 As[2][128 * 32];
  __shared__ __bf16 Bs[2][128 * 32];
  const int tid = threadIdx.x;
  const int lane = tid & 31;
  const int w = tid >> 5;
  const int wm = w >> 1;           // 0..3 -> 32-row band
  const int wn = w & 1;            // 0..1 -> 64-col band
  const int lm = lane & 15;
  const int hs = (lane < 16) ? 0 : 8;
  const int mBlk = blockIdx.x, nBlk = blockIdx.y;
  const int ldRow = tid >> 1;            // 0..127
  const int ldOff = (tid & 1) << 4;      // 0 or 16 halfwords
  const int ldIdx = ldRow * 32 + ldOff;

  v8f acc[2][4] = {};
  const size_t aRowBase = (size_t)(mBlk * 128 + ldRow) * p.K;
  const size_t bRowBase = (size_t)(nBlk * 128 + ldRow) * p.K;

  auto compute = [&](int cb) {
    v16bf af[2], bfm[4];
#pragma unroll
    for (int mt = 0; mt < 2; ++mt) af[mt] = frag_ld(&As[cb][(wm * 32 + mt * 16 + lm) * 32], hs);
#pragma unroll
    for (int nt = 0; nt < 4; ++nt) bfm[nt] = frag_ld(&Bs[cb][(wn * 64 + nt * 16 + lm) * 32], hs);
#pragma unroll
    for (int mt = 0; mt < 2; ++mt)
#pragma unroll
      for (int nt = 0; nt < 4; ++nt)
        acc[mt][nt] = wmma_bf16(af[mt], bfm[nt], acc[mt][nt]);
  };

  // prologue: stage K-step 0 into buffer 0
  {
    const __bf16* ga = p.A + aRowBase + ldOff;
    const __bf16* gb = p.Bt + bRowBase + ldOff;
    v8bf ra0 = *(const v8bf*)ga;
    v8bf ra1 = *(const v8bf*)(ga + 8);
    v8bf rb0 = *(const v8bf*)gb;
    v8bf rb1 = *(const v8bf*)(gb + 8);
    *(v8bf*)&As[0][ldIdx]     = ra0;
    *(v8bf*)&As[0][ldIdx + 8] = ra1;
    *(v8bf*)&Bs[0][ldIdx]     = rb0;
    *(v8bf*)&Bs[0][ldIdx + 8] = rb1;
  }
  __syncthreads();

  int cur = 0;
  for (int k0 = 32; k0 < p.K; k0 += 32) {
    // issue next tile's global loads (fly while WMMAs run)
    const __bf16* ga = p.A + aRowBase + k0 + ldOff;
    const __bf16* gb = p.Bt + bRowBase + k0 + ldOff;
    v8bf ra0 = *(const v8bf*)ga;
    v8bf ra1 = *(const v8bf*)(ga + 8);
    v8bf rb0 = *(const v8bf*)gb;
    v8bf rb1 = *(const v8bf*)(gb + 8);
    __builtin_prefetch(ga + 32, 0, 1);
    __builtin_prefetch(gb + 32, 0, 1);

    // compute on current buffer while loads are in flight
    compute(cur);

    // stage next tile into the other buffer (no reader touches it yet)
    const int nxt = cur ^ 1;
    *(v8bf*)&As[nxt][ldIdx]     = ra0;
    *(v8bf*)&As[nxt][ldIdx + 8] = ra1;
    *(v8bf*)&Bs[nxt][ldIdx]     = rb0;
    *(v8bf*)&Bs[nxt][ldIdx + 8] = rb1;
    __syncthreads();   // single barrier per K-step: cur reads done, nxt writes done
    cur = nxt;
  }
  compute(cur);

  // ---- fused epilogue ----
#pragma unroll
  for (int mt = 0; mt < 2; ++mt)
#pragma unroll
    for (int nt = 0; nt < 4; ++nt)
#pragma unroll
      for (int r = 0; r < 8; ++r) {
        const int row = mBlk * 128 + wm * 32 + mt * 16 + r + hs;
        const int col = nBlk * 128 + wn * 64 + nt * 16 + lm;
        const float v = acc[mt][nt][r];
        switch (p.mode) {
          case EP_EMBED: {
            const int b = row / NPATCH, pp = row % NPATCH;
            const size_t hi = ((size_t)(b * TOKN + pp + 1)) * DIM + col;
            p.outF[hi] = v + p.bias[col] + p.pos[(pp + 1) * DIM + col];
          } break;
          case EP_QKV: {
            const int b = row / TOKN, t = row % TOKN;
            const int sec = col >> 9, rem = col & 511, hh = rem >> 6, d = rem & 63;
            const size_t bh = (size_t)(b * HEADS + hh);
            const __bf16 bv = (__bf16)v;
            if (sec == 0)      p.q[(bh * 64 + t) * 64 + d] = bv;
            else if (sec == 1) p.k[(bh * 64 + t) * 64 + d] = bv;
            else               p.vT[(bh * 64 + d) * 64 + t] = bv;
          } break;
          case EP_BIAS_RES: {
            const size_t idx = (size_t)row * p.N + col;
            p.outF[idx] = v + p.bias[col] + p.res[idx];
          } break;
          case EP_GELU: {
            const size_t idx = (size_t)row * p.N + col;
            p.outB[idx] = (__bf16)gelu_exact(v + p.bias[col]);
          } break;
        }
      }
}

// ---------------- attention: one block per (batch, head), 4 waves, T padded to 64 ----------------
__global__ __launch_bounds__(128) void k_attn(const __bf16* __restrict__ q, const __bf16* __restrict__ k,
                                              const __bf16* __restrict__ vT, __bf16* __restrict__ o) {
  __shared__ float  S[4][16][64];
  __shared__ __bf16 P[4][16][64];
  const int bh = blockIdx.x;
  const int w = threadIdx.x >> 5, lane = threadIdx.x & 31;
  const int lm = lane & 15;
  const int hs = (lane < 16) ? 0 : 8;
  const __bf16* qb = q + (size_t)bh * 4096;
  const __bf16* kb = k + (size_t)bh * 4096;
  const __bf16* vb = vT + (size_t)bh * 4096;

  v8f sa[4] = {};
#pragma unroll
  for (int k0 = 0; k0 < 64; k0 += 32) {
    v16bf aq = frag_ld(qb + (size_t)(w * 16 + lm) * 64 + k0, hs);
#pragma unroll
    for (int nt = 0; nt < 4; ++nt) {
      v16bf bk = frag_ld(kb + (size_t)(nt * 16 + lm) * 64 + k0, hs);
      sa[nt] = wmma_bf16(aq, bk, sa[nt]);
    }
  }
#pragma unroll
  for (int nt = 0; nt < 4; ++nt)
#pragma unroll
    for (int r = 0; r < 8; ++r)
      S[w][r + hs][nt * 16 + lm] = sa[nt][r] * 0.125f;  // SCALE = 64^-0.5

  if (lane < 16) {  // one lane per row of the strip
    float mx = -3.0e38f;
    for (int c = 0; c < TOKN; ++c) mx = fmaxf(mx, S[w][lane][c]);
    float sum = 0.0f;
    for (int c = 0; c < TOKN; ++c) { float e = __expf(S[w][lane][c] - mx); S[w][lane][c] = e; sum += e; }
    const float inv = 1.0f / sum;
    for (int c = 0; c < TOKN; ++c) P[w][lane][c] = (__bf16)(S[w][lane][c] * inv);
    for (int c = TOKN; c < 64; ++c) P[w][lane][c] = (__bf16)0.0f;
  }

  v8f oa[4] = {};
#pragma unroll
  for (int k0 = 0; k0 < 64; k0 += 32) {
    v16bf ap = frag_ld(&P[w][lm][k0], hs);
#pragma unroll
    for (int nt = 0; nt < 4; ++nt) {
      v16bf bv = frag_ld(vb + (size_t)(nt * 16 + lm) * 64 + k0, hs);
      oa[nt] = wmma_bf16(ap, bv, oa[nt]);
    }
  }
  const int b = bh >> 3, hh = bh & 7;
#pragma unroll
  for (int nt = 0; nt < 4; ++nt)
#pragma unroll
    for (int r = 0; r < 8; ++r) {
      const int t = w * 16 + r + hs;
      if (t < TOKN)
        o[((size_t)(b * TOKN + t)) * DIM + hh * 64 + nt * 16 + lm] = (__bf16)oa[nt][r];
    }
}

// ---------------- mean-pool + head LN + classifier ----------------
__global__ __launch_bounds__(256) void k_head(const float* __restrict__ h, const float* __restrict__ g,
                                              const float* __restrict__ bt, const float* __restrict__ Wh,
                                              const float* __restrict__ bh, float* __restrict__ out) {
  __shared__ float pn[512];
  __shared__ float red[18];
  const int b = blockIdx.x;
  const int tid = threadIdx.x;
  const int w = tid >> 5, lane = tid & 31;
  float a0 = 0.f, a1 = 0.f;
  for (int t = 0; t < TOKN; ++t) {
    const float* hr = h + ((size_t)(b * TOKN + t)) * DIM;
    a0 += hr[tid]; a1 += hr[tid + 256];
  }
  a0 *= (1.0f / 57.0f); a1 *= (1.0f / 57.0f);
  pn[tid] = a0; pn[tid + 256] = a1;
  float s = a0 + a1, s2 = a0 * a0 + a1 * a1;
#pragma unroll
  for (int m = 16; m >= 1; m >>= 1) { s += __shfl_xor(s, m, 32); s2 += __shfl_xor(s2, m, 32); }
  if (lane == 0) { red[w] = s; red[8 + w] = s2; }
  __syncthreads();
  if (tid == 0) {
    float S = 0.f, S2 = 0.f;
    for (int i = 0; i < 8; ++i) { S += red[i]; S2 += red[8 + i]; }
    const float mu = S * (1.0f / 512.0f);
    red[16] = mu; red[17] = S2 * (1.0f / 512.0f) - mu * mu;
  }
  __syncthreads();
  const float mu = red[16];
  const float rstd = rsqrtf(red[17] + 1e-5f);
  pn[tid]       = (pn[tid] - mu) * rstd * g[tid] + bt[tid];
  pn[tid + 256] = (pn[tid + 256] - mu) * rstd * g[tid + 256] + bt[tid + 256];
  __syncthreads();
  if (tid < NCLS) {
    float acc = bh[tid];
    for (int d = 0; d < DIM; ++d) acc += pn[d] * Wh[d * NCLS + tid];
    out[b * NCLS + tid] = acc;
  }
}

// ---------------- workspace layout (all sizes multiples of 256 bytes) ----------------
static constexpr size_t SZ_H    = (size_t)ROWS * DIM * 4;          // 59,768,832
static constexpr size_t SZ_Y    = (size_t)ROWS * DIM * 2;          // 29,884,416
static constexpr size_t SZ_O    = SZ_Y;
static constexpr size_t SZ_QKV1 = (size_t)BATCH * HEADS * 64 * 64 * 2;  // 33,554,432 each
static constexpr size_t SZ_U    = (size_t)ROWS * MLPD * 2;         // 119,537,664
static constexpr size_t SZ_POS  = (size_t)TOKN * DIM * 4;
static constexpr size_t SZ_WET  = (size_t)CHN * DIM * 2;
static constexpr size_t SZ_WQ   = (size_t)DEPTH * DIM * 3 * DIM * 2;
static constexpr size_t SZ_WO   = (size_t)DEPTH * DIM * DIM * 2;
static constexpr size_t SZ_W1   = (size_t)DEPTH * DIM * MLPD * 2;
static constexpr size_t SZ_W2   = SZ_W1;

static constexpr size_t OFF_H   = 0;
static constexpr size_t OFF_Y   = OFF_H + SZ_H;
static constexpr size_t OFF_O   = OFF_Y + SZ_Y;
static constexpr size_t OFF_Q   = OFF_O + SZ_O;
static constexpr size_t OFF_K   = OFF_Q + SZ_QKV1;
static constexpr size_t OFF_V   = OFF_K + SZ_QKV1;
static constexpr size_t OFF_U   = OFF_V + SZ_QKV1;   // tok aliases u (disjoint lifetimes)
static constexpr size_t OFF_POS = OFF_U + SZ_U;
static constexpr size_t OFF_WET = OFF_POS + SZ_POS;
static constexpr size_t OFF_WQ  = OFF_WET + SZ_WET;
static constexpr size_t OFF_WOT = OFF_WQ + SZ_WQ;
static constexpr size_t OFF_W1T = OFF_WOT + SZ_WO;
static constexpr size_t OFF_W2T = OFF_W1T + SZ_W1;

extern "C" void kernel_launch(void* const* d_in, const int* in_sizes, int n_in,
                              void* d_out, int out_size, void* d_ws, size_t ws_size,
                              hipStream_t stream) {
  (void)in_sizes; (void)n_in; (void)out_size; (void)ws_size;
  const float* x     = (const float*)d_in[0];
  const float* We    = (const float*)d_in[1];
  const float* be    = (const float*)d_in[2];
  const float* cls   = (const float*)d_in[3];
  const float* Wqkv  = (const float*)d_in[4];
  const float* Wo    = (const float*)d_in[5];
  const float* bo    = (const float*)d_in[6];
  const float* ln1g  = (const float*)d_in[7];
  const float* ln1b  = (const float*)d_in[8];
  const float* W1    = (const float*)d_in[9];
  const float* b1    = (const float*)d_in[10];
  const float* W2    = (const float*)d_in[11];
  const float* b2    = (const float*)d_in[12];
  const float* ln2g  = (const float*)d_in[13];
  const float* ln2b  = (const float*)d_in[14];
  const float* lnh_g = (const float*)d_in[15];
  const float* lnh_b = (const float*)d_in[16];
  const float* Wh    = (const float*)d_in[17];
  const float* bhv   = (const float*)d_in[18];
  float* out = (float*)d_out;

  char* ws = (char*)d_ws;
  float*  h    = (float*)(ws + OFF_H);
  __bf16* y    = (__bf16*)(ws + OFF_Y);
  __bf16* ob   = (__bf16*)(ws + OFF_O);
  __bf16* qb   = (__bf16*)(ws + OFF_Q);
  __bf16* kb   = (__bf16*)(ws + OFF_K);
  __bf16* vTb  = (__bf16*)(ws + OFF_V);
  __bf16* u    = (__bf16*)(ws + OFF_U);
  __bf16* tok  = (__bf16*)(ws + OFF_U);   // alias: tok dead before u is written
  float*  POS  = (float*)(ws + OFF_POS);
  __bf16* WeT  = (__bf16*)(ws + OFF_WET);
  __bf16* WqkvT= (__bf16*)(ws + OFF_WQ);
  __bf16* WoT  = (__bf16*)(ws + OFF_WOT);
  __bf16* W1T  = (__bf16*)(ws + OFF_W1T);
  __bf16* W2T  = (__bf16*)(ws + OFF_W2T);

  // ---- weight prep: f32 -> bf16 transposed [N][K] ----
  auto launch_tr = [&](const float* in, __bf16* outp, int R, int C) {
    k_transpose_bf16<<<(R * C + 255) / 256, 256, 0, stream>>>(in, outp, R, C);
  };
  launch_tr(We, WeT, CHN, DIM);
  for (int l = 0; l < DEPTH; ++l) {
    launch_tr(Wqkv + (size_t)l * DIM * 3 * DIM, WqkvT + (size_t)l * 3 * DIM * DIM, DIM, 3 * DIM);
    launch_tr(Wo   + (size_t)l * DIM * DIM,     WoT   + (size_t)l * DIM * DIM,     DIM, DIM);
    launch_tr(W1   + (size_t)l * DIM * MLPD,    W1T   + (size_t)l * MLPD * DIM,    DIM, MLPD);
    launch_tr(W2   + (size_t)l * MLPD * DIM,    W2T   + (size_t)l * DIM * MLPD,    MLPD, DIM);
  }

  // ---- positional table, cls row, qkv padding zero ----
  k_pos<<<(TOKN * DIM + 255) / 256, 256, 0, stream>>>(POS);
  k_cls<<<(BATCH * DIM + 255) / 256, 256, 0, stream>>>(cls, POS, h);
  {
    size_t n64 = (3 * SZ_QKV1) / 8;
    k_zero64<<<(unsigned)((n64 + 255) / 256), 256, 0, stream>>>((uint64_t*)(ws + OFF_Q), n64);
  }

  // ---- spiral rings -> patch tokens ----
  k_spiral<<<(BATCH * CHN) / 64, 64, 0, stream>>>(x, tok);

  // ---- embedding GEMM: tok[28672,256] @ We -> h rows 1..56 (+be +POS) ----
  {
    GemmArgs g{};
    g.A = tok; g.Bt = WeT; g.bias = be; g.pos = POS; g.outF = h;
    g.N = DIM; g.K = CHN; g.mode = EP_EMBED;
    k_gemm<<<dim3(EROWS / 128, DIM / 128), 256, 0, stream>>>(g);
  }

  // ---- transformer layers ----
  for (int l = 0; l < DEPTH; ++l) {
    k_ln<<<ROWS / 8, 256, 0, stream>>>(h, ln1g + l * DIM, ln1b + l * DIM, y);
    {
      GemmArgs g{};
      g.A = y; g.Bt = WqkvT + (size_t)l * 3 * DIM * DIM;
      g.q = qb; g.k = kb; g.vT = vTb;
      g.N = 3 * DIM; g.K = DIM; g.mode = EP_QKV;
      k_gemm<<<dim3(ROWS / 128, (3 * DIM) / 128), 256, 0, stream>>>(g);
    }
    k_attn<<<BATCH * HEADS, 128, 0, stream>>>(qb, kb, vTb, ob);
    {
      GemmArgs g{};
      g.A = ob; g.Bt = WoT + (size_t)l * DIM * DIM;
      g.bias = bo + l * DIM; g.res = h; g.outF = h;
      g.N = DIM; g.K = DIM; g.mode = EP_BIAS_RES;
      k_gemm<<<dim3(ROWS / 128, DIM / 128), 256, 0, stream>>>(g);
    }
    k_ln<<<ROWS / 8, 256, 0, stream>>>(h, ln2g + l * DIM, ln2b + l * DIM, y);
    {
      GemmArgs g{};
      g.A = y; g.Bt = W1T + (size_t)l * MLPD * DIM;
      g.bias = b1 + l * MLPD; g.outB = u;
      g.N = MLPD; g.K = DIM; g.mode = EP_GELU;
      k_gemm<<<dim3(ROWS / 128, MLPD / 128), 256, 0, stream>>>(g);
    }
    {
      GemmArgs g{};
      g.A = u; g.Bt = W2T + (size_t)l * DIM * MLPD;
      g.bias = b2 + l * DIM; g.res = h; g.outF = h;
      g.N = DIM; g.K = MLPD; g.mode = EP_BIAS_RES;
      k_gemm<<<dim3(ROWS / 128, DIM / 128), 256, 0, stream>>>(g);
    }
  }

  // ---- mean pool + head LN + classifier ----
  k_head<<<BATCH, 256, 0, stream>>>(h, lnh_g, lnh_b, Wh, bhv, out);
}
